// ABE_loss_15539191677792
// MI455X (gfx1250) — matrix operand order repaired and checked by reading, not compiled, not run on hardware
//
#include <hip/hip_runtime.h>

typedef __attribute__((ext_vector_type(16))) __bf16 v16bf;
typedef __attribute__((ext_vector_type(8)))  float  v8f;

#define M_BRANCH   8
#define D_DIM      512
#define BM         128
#define BN         128
#define BK         32
#define LDSS       40        // padded LDS row stride (bf16 elements) -> 80B rows, 16B aligned
#define MARGIN_C   0.5f
#define MARGIN_DIV 0.2f
#define LAMBDA_DIV 0.05f

// ---------- bf16 split helpers (round-to-nearest-even) ----------
__device__ __forceinline__ unsigned short f2bf(float x) {
  union { float f; unsigned u; } a; a.f = x;
  unsigned r = (a.u + 0x7FFFu + ((a.u >> 16) & 1u)) >> 16;
  return (unsigned short)r;
}
__device__ __forceinline__ float bf2f(unsigned short h) {
  union { unsigned u; float f; } a; a.u = ((unsigned)h) << 16;
  return a.f;
}

struct BFrag { union { v16bf v; unsigned u[8]; }; };

// A fragment (16x32, MxK): lane l -> M = l&15; K-halves interleaved by lane group.
// VGPR r<4: K = 2r + 8*khalf ; r>=4: K = 16 + 2(r-4) + 8*khalf   (ISA 7.12.2)
__device__ __forceinline__ void load_fragA(const unsigned short* lds, int row, int khalf, BFrag& f) {
  const unsigned* p = (const unsigned*)(lds + row * LDSS);
#pragma unroll
  for (int r = 0; r < 8; ++r) {
    int kbase = ((r < 4) ? 2 * r : 16 + 2 * (r - 4)) + 8 * khalf;
    f.u[r] = p[kbase >> 1];
  }
}
// B fragment (32x16, KxN): lane l -> N = l&15; same K interleave as A: r<4: K = 2r + 8*khalf ; r>=4: K = 16 + 2(r-4) + 8*khalf.
__device__ __forceinline__ void load_fragB(const unsigned short* lds, int row, int khalf, BFrag& f) {
  const unsigned* p = (const unsigned*)(lds + row * LDSS);
#pragma unroll
  for (int r = 0; r < 8; ++r) {
    int kbase = ((r < 4) ? 2 * r : 16 + 2 * (r - 4)) + 8 * khalf;
    f.u[r] = p[kbase >> 1];
  }
}

// ---------- workspace zero ----------
__global__ void init_ws_kernel(int* ws_words) {
  int i = threadIdx.x;
  if (i < 81) ws_words[i] = 0;   // hist[64] + branch_sum[8] + branch_valid[8] + div_sum[1]
}

// ---------- class histogram (targets in [0,64)) ----------
__global__ void hist_kernel(const int* __restrict__ target, int* __restrict__ hist, int N) {
  int i = blockIdx.x * blockDim.x + threadIdx.x;
  if (i < N) atomicAdd(&hist[target[i]], 1);
}

// ---------- one-time fp32 -> (hi,lo) bf16 split into workspace ----------
__global__ void __launch_bounds__(256)
split_kernel(const float* __restrict__ x, unsigned short* __restrict__ xhi,
             unsigned short* __restrict__ xlo, size_t elems) {
  size_t i = ((size_t)blockIdx.x * blockDim.x + threadIdx.x) * 8;
  if (i + 8 > elems) return;
  float4 a = ((const float4*)(x + i))[0];
  float4 b = ((const float4*)(x + i))[1];
  float f[8] = {a.x, a.y, a.z, a.w, b.x, b.y, b.z, b.w};
  union { unsigned short s[8]; uint4 q; } h, l;
#pragma unroll
  for (int e = 0; e < 8; ++e) {
    h.s[e] = f2bf(f[e]);
    l.s[e] = f2bf(f[e] - bf2f(h.s[e]));
  }
  *(uint4*)(xhi + i) = h.q;
  *(uint4*)(xlo + i) = l.q;
}

// ---------- fused Gram-GEMM + masked reductions ----------
template <bool PRESPLIT>
__global__ void __launch_bounds__(256)
contrastive_kernel(const float* __restrict__ x,
                   const unsigned short* __restrict__ xhi,
                   const unsigned short* __restrict__ xlo,
                   const int* __restrict__ target,
                   const int* __restrict__ hist,
                   float* __restrict__ branch_sum,
                   int* __restrict__ branch_valid, int N) {
  __shared__ unsigned short Ah[BM * LDSS], Al[BM * LDSS];
  __shared__ unsigned short Bh[BN * LDSS], Bl[BN * LDSS];
  __shared__ int tgt[BN];

  const int m        = blockIdx.y;
  const int rowBaseN = blockIdx.x * BM;
  const size_t mbase = (size_t)m * N * D_DIM;
  const float* xb    = x + mbase;

  const int tid  = threadIdx.x;
  const int lane = tid & 31;
  const int w    = tid >> 5;
  const int half = lane >> 4;
  const int l15  = lane & 15;
  const int waveRow = rowBaseN + w * 16;   // 16 rows owned by this wave

  // targets for the rows mapped to this lane's accumulator elements
  int trow[8];
#pragma unroll
  for (int e = 0; e < 8; ++e) trow[e] = target[waveRow + half * 8 + e];

  float posc[8], poss[8], negs[8];
#pragma unroll
  for (int e = 0; e < 8; ++e) { posc[e] = 0.f; poss[e] = 0.f; negs[e] = 0.f; }

  const int nJT = N / BN;
  const int nKT = D_DIM / BK;

  const int r  = tid >> 1;          // staging row handled by this thread
  const int c0 = (tid & 1) * 16;    // 16-element half-row segment

  for (int jt = 0; jt < nJT; ++jt) {
    __syncthreads();                // protect tgt/LDS vs previous epilogue
    if (tid < BN) tgt[tid] = target[jt * BN + tid];

    v8f acc[8];
#pragma unroll
    for (int s = 0; s < 8; ++s)
#pragma unroll
      for (int i = 0; i < 8; ++i) acc[s][i] = 0.0f;

    for (int kt = 0; kt < nKT; ++kt) {
      size_t offA = mbase + (size_t)(rowBaseN + r) * D_DIM + kt * BK + c0;
      size_t offB = mbase + (size_t)(jt * BN   + r) * D_DIM + kt * BK + c0;
      __syncthreads();
      if (PRESPLIT) {
        // pure copies: bf16 hi/lo tiles from workspace -> LDS (no VALU conversion)
        const uint4* pAh = (const uint4*)(xhi + offA);
        const uint4* pAl = (const uint4*)(xlo + offA);
        const uint4* pBh = (const uint4*)(xhi + offB);
        const uint4* pBl = (const uint4*)(xlo + offB);
        uint4* dAh = (uint4*)&Ah[r * LDSS + c0];
        uint4* dAl = (uint4*)&Al[r * LDSS + c0];
        uint4* dBh = (uint4*)&Bh[r * LDSS + c0];
        uint4* dBl = (uint4*)&Bl[r * LDSS + c0];
        dAh[0] = pAh[0]; dAh[1] = pAh[1];
        dAl[0] = pAl[0]; dAl[1] = pAl[1];
        dBh[0] = pBh[0]; dBh[1] = pBh[1];
        dBl[0] = pBl[0]; dBl[1] = pBl[1];
        if (kt + 1 < nKT) {
          // pull next k-chunk toward the WGP while this chunk's WMMAs run
          __builtin_prefetch(xhi + offA + BK, 0, 3);
          __builtin_prefetch(xlo + offA + BK, 0, 3);
          __builtin_prefetch(xhi + offB + BK, 0, 3);
          __builtin_prefetch(xlo + offB + BK, 0, 3);
        }
      } else {
        // fallback: load fp32 and split on the fly
        const float4* ga = (const float4*)(xb + (size_t)(rowBaseN + r) * D_DIM + kt * BK + c0);
        const float4* gb = (const float4*)(xb + (size_t)(jt * BN   + r) * D_DIM + kt * BK + c0);
#pragma unroll
        for (int q = 0; q < 4; ++q) {
          float4 va = ga[q];
          float4 vb = gb[q];
          float fa[4] = {va.x, va.y, va.z, va.w};
          float fb[4] = {vb.x, vb.y, vb.z, vb.w};
#pragma unroll
          for (int e2 = 0; e2 < 4; ++e2) {
            int c = c0 + q * 4 + e2;
            unsigned short h = f2bf(fa[e2]);
            Ah[r * LDSS + c] = h;
            Al[r * LDSS + c] = f2bf(fa[e2] - bf2f(h));
            unsigned short h2 = f2bf(fb[e2]);
            Bh[r * LDSS + c] = h2;
            Bl[r * LDSS + c] = f2bf(fb[e2] - bf2f(h2));
          }
        }
      }
      __syncthreads();

      BFrag ah, al;
      load_fragA(Ah, w * 16 + l15, half, ah);
      load_fragA(Al, w * 16 + l15, half, al);
#pragma unroll
      for (int s = 0; s < 8; ++s) {
        BFrag bh, bl;
        load_fragB(Bh, s * 16 + l15, half, bh);
        load_fragB(Bl, s * 16 + l15, half, bl);
        // S = hi*hi + hi*lo + lo*hi  (drop lo*lo; ~2^-16 rel err)
        acc[s] = __builtin_amdgcn_wmma_f32_16x16x32_bf16(false, ah.v, false, bh.v, (short)0, acc[s], false, false);
        acc[s] = __builtin_amdgcn_wmma_f32_16x16x32_bf16(false, ah.v, false, bl.v, (short)0, acc[s], false, false);
        acc[s] = __builtin_amdgcn_wmma_f32_16x16x32_bf16(false, al.v, false, bh.v, (short)0, acc[s], false, false);
      }
    }

    // fused epilogue: branchless masked reductions over this 16x128 strip of S
#pragma unroll
    for (int s = 0; s < 8; ++s) {
      int tj = tgt[s * 16 + l15];
#pragma unroll
      for (int e = 0; e < 8; ++e) {
        float sv   = acc[s][e];
        bool  same = (tj == trow[e]);
        bool  pos  = same && (sv < 1.0f);
        posc[e] += pos ? 1.0f : 0.0f;
        poss[e] += pos ? fmaxf(MARGIN_C - sv, 0.0f) : 0.0f;
        negs[e] += same ? 0.0f : sv;
      }
    }
  }

  // reduce across the 16 lanes of each half-wave (columns of the C tile)
#pragma unroll
  for (int e = 0; e < 8; ++e) {
    for (int off = 1; off < 16; off <<= 1) {
      posc[e] += __shfl_xor(posc[e], off, 32);
      poss[e] += __shfl_xor(poss[e], off, 32);
      negs[e] += __shfl_xor(negs[e], off, 32);
    }
  }

  if (l15 == 0) {   // lane 0 -> rows +0..7, lane 16 -> rows +8..15
    float lsum = 0.0f; int lval = 0;
#pragma unroll
    for (int e = 0; e < 8; ++e) {
      float pc = posc[e];
      int   nc = N - hist[trow[e]];
      float pl = poss[e] / fmaxf(pc, 1.0f);
      float nl = negs[e] / (float)(nc > 1 ? nc : 1);
      if (pc > 0.0f && nc > 0) { lsum += pl + nl; lval += 1; }
    }
    atomicAdd(&branch_sum[m], lsum);
    atomicAdd(&branch_valid[m], lval);
  }
}

// ---------- divergence: 28 branch pairs x N dots of length D ----------
__global__ void __launch_bounds__(256)
divergence_kernel(const float* __restrict__ x, float* __restrict__ div_sum, int N) {
  __shared__ float red[256];
  int p = blockIdx.y;
  int i = 0, j = 1, rem = p;
  for (int a = 0; a < M_BRANCH - 1; ++a) {
    int cnt = M_BRANCH - 1 - a;
    if (rem < cnt) { i = a; j = a + 1 + rem; break; }
    rem -= cnt;
  }
  int n = blockIdx.x * blockDim.x + threadIdx.x;
  float v = 0.0f;
  if (n < N) {
    const float4* xa = (const float4*)(x + ((size_t)i * N + n) * D_DIM);
    const float4* xc = (const float4*)(x + ((size_t)j * N + n) * D_DIM);
    float s = 0.0f;
    for (int d = 0; d < D_DIM / 4; ++d) {
      float4 a4 = xa[d], b4 = xc[d];
      s += a4.x * b4.x + a4.y * b4.y + a4.z * b4.z + a4.w * b4.w;
    }
    v = fmaxf(s - MARGIN_DIV, 0.0f);
  }
  red[threadIdx.x] = v;
  __syncthreads();
  for (int off = 128; off > 0; off >>= 1) {
    if (threadIdx.x < off) red[threadIdx.x] += red[threadIdx.x + off];
    __syncthreads();
  }
  if (threadIdx.x == 0) atomicAdd(div_sum, red[0]);
}

// ---------- final scalar ----------
__global__ void finalize_kernel(const float* branch_sum, const int* branch_valid,
                                const float* div_sum, float* out, int N) {
  float c = 0.0f;
  for (int m = 0; m < M_BRANCH; ++m) {
    int v = branch_valid[m];
    c += branch_sum[m] / (float)(v > 1 ? v : 1);
  }
  c /= (float)M_BRANCH;
  float npairs = (float)(M_BRANCH * (M_BRANCH - 1) / 2);
  float d = div_sum[0] / (npairs * (float)N);
  out[0] = c + LAMBDA_DIV * d;
}

extern "C" void kernel_launch(void* const* d_in, const int* in_sizes, int n_in,
                              void* d_out, int out_size, void* d_ws, size_t ws_size,
                              hipStream_t stream) {
  (void)n_in; (void)out_size;
  const float* x      = (const float*)d_in[0];
  const int*   target = (const int*)d_in[1];
  const int N = in_sizes[1];

  int*   hist         = (int*)d_ws;              // 64 ints
  float* branch_sum   = (float*)d_ws + 64;       // 8 floats
  int*   branch_valid = (int*)d_ws + 72;         // 8 ints
  float* div_sum      = (float*)d_ws + 80;       // 1 float

  const size_t SPLIT_OFF = 4096;                 // bytes; keeps split arrays 16B-aligned
  const size_t elems = (size_t)M_BRANCH * (size_t)N * (size_t)D_DIM;
  const size_t split_bytes = 2 * elems * sizeof(unsigned short);
  const bool presplit = ws_size >= SPLIT_OFF + split_bytes;
  unsigned short* xhi = (unsigned short*)((char*)d_ws + SPLIT_OFF);
  unsigned short* xlo = xhi + elems;

  init_ws_kernel<<<1, 128, 0, stream>>>((int*)d_ws);
  hist_kernel<<<(N + 255) / 256, 256, 0, stream>>>(target, hist, N);

  dim3 grid(N / BM, M_BRANCH);
  if (presplit) {
    size_t nthr = elems / 8;
    split_kernel<<<(unsigned)((nthr + 255) / 256), 256, 0, stream>>>(x, xhi, xlo, elems);
    contrastive_kernel<true><<<grid, 256, 0, stream>>>(x, xhi, xlo, target, hist,
                                                       branch_sum, branch_valid, N);
  } else {
    contrastive_kernel<false><<<grid, 256, 0, stream>>>(x, xhi, xlo, target, hist,
                                                        branch_sum, branch_valid, N);
  }

  dim3 gdiv(N / 256, M_BRANCH * (M_BRANCH - 1) / 2);
  divergence_kernel<<<gdiv, 256, 0, stream>>>(x, div_sum, N);

  finalize_kernel<<<1, 1, 0, stream>>>(branch_sum, branch_valid, div_sum, (float*)d_out, N);
}
